// Attention_23218593202595
// MI455X (gfx1250) — matrix operand order repaired and checked by reading, not compiled, not run on hardware
//
#include <hip/hip_runtime.h>
#include <hip/hip_bf16.h>

// ---------------------------------------------------------------------------
// Attention scoring: scores = softmax_S( v . tanh(W @ [static;dynamic;dec]) )
// B=256, H=256, S=512.
// Main GEMM (K=512, static+dynamic) on v_wmma_f32_16x16x32_bf16, f32 accum,
// double-buffered LDS staging, v_perm_b32 bf16 packing, v_tanh_f32 epilogue.
// Decoder term is s-independent -> tiny per-wave GEMV bias, added pre-tanh.
// ---------------------------------------------------------------------------

typedef __bf16 v16bf __attribute__((ext_vector_type(16)));
typedef float  v8f   __attribute__((ext_vector_type(8)));

constexpr int Bn = 256;
constexpr int Hn = 256;
constexpr int Sn = 512;
constexpr int WSTRIDE = 768;    // W row stride in floats (3H)
constexpr int NT = 64;          // columns (s) per workgroup
constexpr int KSTEP = 32;       // K per WMMA
constexpr int NSTEPS = 16;      // 512 K / 32
constexpr int LDSB_STRIDE = 17; // dword stride (odd -> conflict-free strided reads)

union AFrag {
    unsigned u[8];
    v16bf    v;
};

// f32 pair -> packed bf16 (round-half-up), single v_perm_b32 for the pack.
__device__ __forceinline__ unsigned pack_bf16(float lo, float hi) {
    const unsigned a = __float_as_uint(lo) + 0x8000u;
    const unsigned b = __float_as_uint(hi) + 0x8000u;
    // result = { b[31:16], a[31:16] }
    return __builtin_amdgcn_perm(b, a, 0x07060302u);
}

// CDNA5 hardware tanh (TRANS op). ISA: 1 instruction of separation before the
// result may be consumed -> s_delay_alu TRANS32_DEP_1.
__device__ __forceinline__ float hw_tanh(float x) {
    float y;
    asm volatile("v_tanh_f32 %0, %1\n\t"
                 "s_delay_alu instid0(TRANS32_DEP_1)"
                 : "=v"(y) : "v"(x));
    return y;
}

__launch_bounds__(512)
__global__ void attn_scores_kernel(const float* __restrict__ sh,   // [B,H,S]
                                   const float* __restrict__ dh,   // [B,H,S]
                                   const float* __restrict__ dec,  // [B,H]
                                   const float* __restrict__ W,    // [H,3H]
                                   const float* __restrict__ vw,   // [H]
                                   float* __restrict__ out)        // [B,S] raw scores
{
    __shared__ unsigned lds_b[2][NT * LDSB_STRIDE];
    __shared__ float    lds_score[NT];

    const int t    = threadIdx.x;
    const int wave = t >> 5;        // 0..15 -> M strip [16*wave, 16*wave+16)
    const int lane = t & 31;
    const int b    = blockIdx.y;
    const int s0   = blockIdx.x * NT;

    if (t < NT) lds_score[t] = 0.0f;

    v8f acc[4] = {v8f{}, v8f{}, v8f{}, v8f{}};

    // ---- A-fragment addressing (ISA 16-bit A 16x32 layout) ----
    //   lane L: M = L%16; dwords 0-3 hold K = khalf..khalf+7 (pairs),
    //           dwords 4-7 hold K = khalf+16..khalf+23; khalf = (L<16 ? 0 : 8)
    const int hrow  = wave * 16 + (lane & 15);
    const int khalf = (lane >> 4) << 3;              // 0 or 8
    const float* Wp = W + (size_t)hrow * WSTRIDE + khalf;

    // ---- staging assignment: thread -> (column, k-pair), coalesced along s ----
    const int cstage = t & 63;      // 0..63
    const int kp0    = t >> 6;      // 0..7 (also covers kp0+8)
    const size_t colbase = ((size_t)b * Hn) * Sn + s0 + cstage;
    const float* src0 = sh + colbase + (size_t)(2 * kp0) * Sn;   // static rows
    const float* src1 = dh + colbase + (size_t)(2 * kp0) * Sn;   // dynamic rows
    const int stgoff = cstage * LDSB_STRIDE + kp0;

    // ---- B-fragment base (ISA 16-bit B 32x16 layout) ----
    //   lane L: N = L%16; K-half selected by lane/16 (khalf/8)
    const int bbase = (lane & 15) * LDSB_STRIDE + ((lane >> 4) << 2);

    // stage K-step `step` into LDS buffer `bufi`
    auto stage = [&](int step, int bufi) {
        const float* s = (step < 8)
            ? (src0 + (size_t)step * (KSTEP * Sn))
            : (src1 + (size_t)(step - 8) * (KSTEP * Sn));
        const float f0 = s[0];
        const float f1 = s[Sn];
        const float f2 = s[16 * Sn];
        const float f3 = s[17 * Sn];
        unsigned* p = &lds_b[bufi][stgoff];
        p[0] = pack_bf16(f0, f1);
        p[8] = pack_bf16(f2, f3);
    };

    // ======================= main K loop (double-buffered) ==================
    stage(0, 0);
    for (int step = 0; step < NSTEPS; ++step) {
        __syncthreads();                      // buf[step&1] now visible
        if (step + 1 < NSTEPS) stage(step + 1, (step + 1) & 1);

        // A fragment: W row block (L2-resident), pack to bf16 via v_perm
        AFrag a;
        {
            const float4 g0 = *(const float4*)(Wp);
            const float4 g1 = *(const float4*)(Wp + 4);
            const float4 g2 = *(const float4*)(Wp + 16);
            const float4 g3 = *(const float4*)(Wp + 20);
            a.u[0] = pack_bf16(g0.x, g0.y);  a.u[1] = pack_bf16(g0.z, g0.w);
            a.u[2] = pack_bf16(g1.x, g1.y);  a.u[3] = pack_bf16(g1.z, g1.w);
            a.u[4] = pack_bf16(g2.x, g2.y);  a.u[5] = pack_bf16(g2.z, g2.w);
            a.u[6] = pack_bf16(g3.x, g3.y);  a.u[7] = pack_bf16(g3.z, g3.w);
        }
        Wp += KSTEP;

        // 4 column sub-tiles: LDS fragment + WMMA (reads buf[step&1])
        const unsigned* lb = lds_b[step & 1];
        #pragma unroll
        for (int j = 0; j < 4; ++j) {
            AFrag bf;
            const int base = bbase + j * (16 * LDSB_STRIDE);
            #pragma unroll
            for (int i = 0; i < 8; ++i) bf.u[i] = lb[base + i + (i & 4)];
            acc[j] = __builtin_amdgcn_wmma_f32_16x16x32_bf16(
                         false, a.v, false, bf.v, (short)0, acc[j], false, false);
        }
    }

    // ============== decoder bias: bias[h] = W_dec[h,:] . dec[b,:] ==========
    // lane computes half-row partial (128 MACs), combine with partner lane.
    float bpart = 0.0f;
    {
        const int  koff = (lane >> 4) << 7;   // 0 or 128
        const float* Wd = W + (size_t)hrow * WSTRIDE + 2 * Hn + koff;
        const float* dv = dec + (size_t)b * Hn + koff;
        #pragma unroll 8
        for (int k = 0; k < 128; ++k) bpart += Wd[k] * dv[k];
    }
    bpart += __shfl_xor(bpart, 16, 32);       // full row sum, rows keyed by lane%16

    // redistribute: dword r of acc holds row (wave*16 + r + 8*(lane>=16))
    float bias_r[8];
    #pragma unroll
    for (int r = 0; r < 8; ++r)
        bias_r[r] = __shfl(bpart, r + ((lane >> 4) << 3), 32);

    // ---- epilogue: scores[c] += sum_r v[h_r] * tanh(acc + bias) ----
    float vv[8];
    const int rbase = wave * 16 + ((lane >> 4) << 3);
    #pragma unroll
    for (int r = 0; r < 8; ++r) vv[r] = vw[rbase + r];

    #pragma unroll
    for (int j = 0; j < 4; ++j) {
        float ps = 0.0f;
        #pragma unroll
        for (int r = 0; r < 8; ++r) ps += vv[r] * hw_tanh(acc[j][r] + bias_r[r]);
        atomicAdd(&lds_score[j * 16 + (lane & 15)], ps);
    }
    __syncthreads();

    if (t < NT) out[(size_t)b * Sn + s0 + t] = lds_score[t];
}

// ---------------------------------------------------------------------------
// In-place softmax over S=512 per batch row. One block per row, 512 threads.
// ---------------------------------------------------------------------------
__launch_bounds__(512)
__global__ void softmax_kernel(float* __restrict__ out)
{
    __shared__ float redmax[16];
    __shared__ float redsum[16];

    const int b = blockIdx.x;
    const int t = threadIdx.x;

    const float x = out[(size_t)b * Sn + t];

    float m = x;
    #pragma unroll
    for (int off = 16; off > 0; off >>= 1)
        m = fmaxf(m, __shfl_xor(m, off, 32));
    if ((t & 31) == 0) redmax[t >> 5] = m;
    __syncthreads();

    float mall = redmax[0];
    #pragma unroll
    for (int i = 1; i < 16; ++i) mall = fmaxf(mall, redmax[i]);

    const float e = expf(x - mall);

    float s = e;
    #pragma unroll
    for (int off = 16; off > 0; off >>= 1)
        s += __shfl_xor(s, off, 32);
    if ((t & 31) == 0) redsum[t >> 5] = s;
    __syncthreads();

    float sall = 0.0f;
    #pragma unroll
    for (int i = 0; i < 16; ++i) sall += redsum[i];

    out[(size_t)b * Sn + t] = e / sall;
}

extern "C" void kernel_launch(void* const* d_in, const int* in_sizes, int n_in,
                              void* d_out, int out_size, void* d_ws, size_t ws_size,
                              hipStream_t stream)
{
    const float* sh  = (const float*)d_in[0];  // static_hidden  [B,H,S]
    const float* dh  = (const float*)d_in[1];  // dynamic_hidden [B,H,S]
    const float* dec = (const float*)d_in[2];  // decoder_hidden [B,H]
    const float* W   = (const float*)d_in[3];  // W [1,H,3H]
    const float* vw  = (const float*)d_in[4];  // v [1,1,H]
    float* out = (float*)d_out;                // [B,1,S]

    dim3 grid1(Sn / NT, Bn);
    attn_scores_kernel<<<grid1, 512, 0, stream>>>(sh, dh, dec, W, vw, out);
    softmax_kernel<<<Bn, Sn, 0, stream>>>(out);
}
